// PointNetSWFPModule_65953517797927
// MI455X (gfx1250) — matrix operand order repaired and checked by reading, not compiled, not run on hardware
//
#include <hip/hip_runtime.h>
#include <hip/hip_bf16.h>

typedef __attribute__((ext_vector_type(2)))  float  v2f;
typedef __attribute__((ext_vector_type(8)))  float  v8f;
typedef __attribute__((ext_vector_type(16))) __bf16 v16bf;
typedef __attribute__((ext_vector_type(4)))  unsigned int u32x4;
typedef __attribute__((ext_vector_type(8)))  int i32x8;
typedef __attribute__((ext_vector_type(4)))  int i32x4;

#define AS1 __attribute__((address_space(1)))
#define AS3 __attribute__((address_space(3)))

#define B_    4
#define N_    16384
#define M_    4096
#define K_    8
#define C_    64
#define OUT_  128
#define P1_   (B_*N_*K_)   /* 524288 positions (sim MLP) */
#define P2_   (B_*N_)      /* 65536  positions (final MLP) */
#define G1_BLOCKS (P1_/128)  /* 4096 */
#define G2_BLOCKS (P2_/128)  /* 512  */

#if defined(__AMDGCN__) && __has_builtin(__builtin_amdgcn_wmma_f32_16x16x4_f32)
#define HAVE_WMMA_F32X4 1
#else
#define HAVE_WMMA_F32X4 0
#endif

#if defined(__AMDGCN__) && __has_builtin(__builtin_amdgcn_tensor_load_to_lds) && __has_builtin(__builtin_amdgcn_s_wait_tensorcnt)
#define HAVE_TDM 1
#else
#define HAVE_TDM 0
#endif

#if defined(__AMDGCN__) && __has_builtin(__builtin_amdgcn_global_load_async_to_lds_b128) && __has_builtin(__builtin_amdgcn_s_wait_asynccnt)
#define HAVE_ASYNCLDS 1
#else
#define HAVE_ASYNCLDS 0
#endif

// ---------------------------------------------------------------- helpers
__device__ __forceinline__ void topk_insert(float bd[8], int bi[8], float d, int ci) {
  float cd = d; int cx = ci;
#pragma unroll
  for (int j = 0; j < 8; ++j) {
    bool sw = cd < bd[j];
    float td = bd[j]; int ti = bi[j];
    bd[j] = sw ? cd : td;  bi[j] = sw ? cx : ti;
    cd   = sw ? td : cd;   cx   = sw ? ti : cx;
  }
}

// ---------------------------------------------------------------- pack centers: (cx,cy,cz,|c|^2)
__global__ __launch_bounds__(256) void pack_centers_k(const float* __restrict__ cc,
                                                      float* __restrict__ packedC) {
  int i = blockIdx.x * 256 + threadIdx.x;            // b*M + m
  int b = i / M_, m = i - b * M_;
  float x = cc[(b*3 + 0) * M_ + m];
  float y = cc[(b*3 + 1) * M_ + m];
  float z = cc[(b*3 + 2) * M_ + m];
  packedC[(size_t)i*4 + 0] = x;
  packedC[(size_t)i*4 + 1] = y;
  packedC[(size_t)i*4 + 2] = z;
  packedC[(size_t)i*4 + 3] = x*x + y*y + z*z;
}

// ---------------------------------------------------------------- kNN via V_WMMA_F32_16X16X4_F32
// Workgroup = 8 waves = 128 consecutive points of one batch. The 64 KB packed
// center table for that batch is staged in LDS once (TDM if available, else
// async-load-to-LDS, else plain copy), then each wave runs 256 f32 WMMAs
// against LDS-resident A-fragments and maintains a per-lane top-8 with a
// branch-guarded insertion (wave-coherent skip once the list stabilizes).
// D layout: lane l -> point col n = l&15, center rows m = r + 8*(l>=16).
__global__ __launch_bounds__(256) void knn_k(const float* __restrict__ pc,
                                             const float* __restrict__ packedC,
                                             int* __restrict__ idxw,
                                             float* __restrict__ out_idx) {
  extern __shared__ float ldsC[];                    // M_*4 floats = 64 KB
  const int b = blockIdx.x >> 7;                     // 128 blocks per batch

  // ---- stage per-batch packed centers into LDS ----
#if HAVE_TDM
  if ((threadIdx.x >> 5) == 0) {
    unsigned long long ga = (unsigned long long)(packedC + (size_t)b * M_ * 4);
    u32x4 g0;
    g0[0] = 1u;                                       // count=1 (valid), user mode
    g0[1] = 0u;                                       // lds_addr = 0 (dynamic LDS base)
    g0[2] = (unsigned int)(ga & 0xffffffffu);         // global_addr[31:0]
    g0[3] = (unsigned int)((ga >> 32) & 0x1ffffffu) | (2u << 30);  // addr[56:32] | type=2
    i32x8 g1;
    g1[0] = (2 << 16);                                // data_size = 4 bytes
    g1[1] = (int)(16384u << 16);                      // tensor_dim0[15:0]=16384 elems
    g1[2] = (1 << 16);                                // tensor_dim0[31:16]=0 | tensor_dim1=1
    g1[3] = (int)(16384u << 16);                      // tile_dim0 = 16384
    g1[4] = 1;                                        // tile_dim1 = 1, tile_dim2 = 0
    g1[5] = 16384;                                    // tensor_dim0_stride
    g1[6] = 0;
    g1[7] = 0;
    i32x4 gz;  gz[0] = gz[1] = gz[2] = gz[3] = 0;
    i32x8 gz8;
#pragma unroll
    for (int z = 0; z < 8; ++z) gz8[z] = 0;
#if __clang_major__ >= 23
    __builtin_amdgcn_tensor_load_to_lds(g0, g1, gz, gz, gz8, 0);   // 6-arg toolchain
#else
    __builtin_amdgcn_tensor_load_to_lds(g0, g1, gz, gz, 0);        // 5-arg toolchain
#endif
    __builtin_amdgcn_s_wait_tensorcnt(0);
  }
  __syncthreads();
#elif HAVE_ASYNCLDS
  {
    const float* gsrc = packedC + (size_t)b * M_ * 4;
    for (int i = threadIdx.x; i < M_; i += 256) {    // 16 B per lane per iter
      __builtin_amdgcn_global_load_async_to_lds_b128(
          (AS1 void*)(gsrc + (size_t)i * 4),
          (AS3 void*)(ldsC + (size_t)i * 4), 0, 0);
    }
    __builtin_amdgcn_s_wait_asynccnt(0);
  }
  __syncthreads();
#else
  {
    const float4* gsrc = (const float4*)(packedC + (size_t)b * M_ * 4);
    float4* dst = (float4*)ldsC;
    for (int i = threadIdx.x; i < M_; i += 256) dst[i] = gsrc[i];
  }
  __syncthreads();
#endif

  const int wave = blockIdx.x * 8 + (threadIdx.x >> 5);
  const int lane = threadIdx.x & 31;
  const int n0 = (wave - b * (N_/16)) * 16;
  const int nl = n0 + (lane & 15);
  const int ml = lane & 15;

  float px = pc[(b*3 + 0) * N_ + nl];
  float py = pc[(b*3 + 1) * N_ + nl];
  float pz = pc[(b*3 + 2) * N_ + nl];
  float bv[4] = { -2.f*px, -2.f*py, -2.f*pz, 1.f };
  const int koff  = (lane >> 4) << 1;   // 0 or 2
  const int half8 = (lane >> 4) << 3;   // 0 or 8

#if HAVE_WMMA_F32X4
  v2f bfrag; bfrag.x = bv[koff]; bfrag.y = bv[koff + 1];
  v8f dzero;
#pragma unroll
  for (int r = 0; r < 8; ++r) dzero[r] = 0.f;
#endif

  float bd[8]; int bi[8];
#pragma unroll
  for (int j = 0; j < 8; ++j) { bd[j] = 3.0e38f; bi[j] = 0; }

  for (int ct = 0; ct < M_/16; ++ct) {
#if HAVE_WMMA_F32X4
    {
      const float* cp = ldsC + (((ct*16 + ml) << 2) + koff);
      v2f afrag; afrag.x = cp[0]; afrag.y = cp[1];
      v8f dist = __builtin_amdgcn_wmma_f32_16x16x4_f32(
          false, afrag, false, bfrag, (short)0, dzero, false, false);
#pragma unroll
      for (int r = 0; r < 8; ++r) {
        float d = dist[r];
        if (d < bd[7]) topk_insert(bd, bi, d, ct*16 + half8 + r);
      }
    }
#else
    {
#pragma unroll
      for (int r = 0; r < 8; ++r) {
        int ci = ct*16 + half8 + r;
        const float* cp = ldsC + (ci << 2);
        float d = cp[3] + bv[0]*cp[0] + bv[1]*cp[1] + bv[2]*cp[2];
        if (d < bd[7]) topk_insert(bd, bi, d, ci);
      }
    }
#endif
  }

  // merge the two half-wave lists (each half saw half the centers)
  float od[8]; int oi[8];
#pragma unroll
  for (int j = 0; j < 8; ++j) {
    od[j] = __shfl_xor(bd[j], 16, 32);
    oi[j] = __shfl_xor(bi[j], 16, 32);
  }
#pragma unroll
  for (int j = 0; j < 8; ++j) topk_insert(bd, bi, od[j], oi[j]);

  if (lane < 16) {
    size_t base = ((size_t)b * N_ + nl) * K_;
#pragma unroll
    for (int j = 0; j < 8; ++j) {
      idxw[base + j]    = bi[j];
      out_idx[base + j] = (float)bi[j];
    }
  }
}

// ---------------------------------------------------------------- sim MLP layer1: h = W1 @ [f0;f1] + b1
// GEMM [64 x 128] x [128 x P1], gather fused into B-fragment build, bf16 WMMA, f32 acc.
__global__ __launch_bounds__(256) void gemm1_k(const float* __restrict__ pf,
                                               const float* __restrict__ w1,
                                               const float* __restrict__ b1,
                                               const int* __restrict__ idxw,
                                               float* __restrict__ hbuf,
                                               float* __restrict__ part1) {
  __shared__ __bf16 lw[C_ * 2*C_];           // 64x128 bf16 weights
  __shared__ float lsum[8][C_], lsq[8][C_];  // per-wave deterministic partials

  int tid = threadIdx.x;
  for (int i = tid; i < C_*2*C_; i += 256) lw[i] = (__bf16)w1[i];
  __syncthreads();

  const int wave = tid >> 5, lane = tid & 31;
  const int p   = (blockIdx.x * 8 + wave) * 16 + (lane & 15);
  const int b   = p / (N_*K_);
  const int rem = p - b * (N_*K_);
  const int n   = rem / K_;
  const int g   = idxw[p];                    // center index (reference gathers pf at it)
  const int coff  = (lane >> 4) << 4;
  const int half8 = (lane >> 4) << 3;
  const int ml    = lane & 15;

  v8f acc[4];
#pragma unroll
  for (int t = 0; t < 4; ++t)
#pragma unroll
    for (int r = 0; r < 8; ++r) acc[t][r] = 0.f;

#pragma unroll
  for (int s = 0; s < 4; ++s) {
    v16bf bfrag;
#pragma unroll
    for (int j = 0; j < 16; ++j) {
      int c = s*32 + half8 + (j & 7) + ((j >> 3) << 4);
      float v = (c < C_) ? pf[((size_t)b*C_ + c      ) * N_ + g]
                         : pf[((size_t)b*C_ + (c-C_)) * N_ + n];
      bfrag[j] = (__bf16)v;
    }
#pragma unroll
    for (int t = 0; t < 4; ++t) {
      v16bf afrag;
#pragma unroll
      for (int j = 0; j < 16; ++j) {
        int i = j >> 1, pb = j & 1;
        int kl = ((i >> 2) << 4) + half8 + ((i & 3) << 1) + pb;
        afrag[j] = lw[(t*16 + ml) * (2*C_) + s*32 + kl];
      }
      acc[t] = __builtin_amdgcn_wmma_f32_16x16x32_bf16(
          false, afrag, false, bfrag, (short)0, acc[t], false, false);
    }
  }

#pragma unroll
  for (int t = 0; t < 4; ++t) {
#pragma unroll
    for (int r = 0; r < 8; ++r) {
      int o = t*16 + half8 + r;
      float v = acc[t][r] + b1[o];
      hbuf[(size_t)o * P1_ + p] = v;
      float s1 = v, s2 = v*v;
#pragma unroll
      for (int m = 1; m < 16; m <<= 1) { s1 += __shfl_xor(s1, m, 32); s2 += __shfl_xor(s2, m, 32); }
      if (ml == 0) { lsum[wave][o] = s1; lsq[wave][o] = s2; }
    }
  }
  __syncthreads();
  if (tid < C_) {
    float a = 0.f, q = 0.f;
#pragma unroll
    for (int w = 0; w < 8; ++w) { a += lsum[w][tid]; q += lsq[w][tid]; }
    part1[(size_t)blockIdx.x * (2*C_) + tid]      = a;
    part1[(size_t)blockIdx.x * (2*C_) + C_ + tid] = q;
  }
}

// ---------------------------------------------------------------- fixed-order stats reduction
__global__ void reduce_stats_k(const float* __restrict__ part, float* __restrict__ stats,
                               int nblk, int width) {
  int c = threadIdx.x;
  if (c < width) {
    float a = 0.f;
    for (int i = 0; i < nblk; ++i) a += part[(size_t)i * width + c];
    stats[c] = a;
  }
}

// ---------------------------------------------------------------- fused: BN + ReLU + layer2 + sigmoid -> w,
// then kNN-weighted interpolation of center features (one thread per point).
__global__ __launch_bounds__(256) void wint_k(const float* __restrict__ hbuf,
                                              const float* __restrict__ stats1,
                                              const float* __restrict__ g1,
                                              const float* __restrict__ be1,
                                              const float* __restrict__ w2,
                                              const float* __restrict__ b2,
                                              const float* __restrict__ cf,
                                              const int* __restrict__ idxw,
                                              float* __restrict__ out_w,
                                              float* __restrict__ interp) {
  __shared__ float sc[C_], sh[C_], sw2[C_];
  int tid = threadIdx.x;
  if (tid < C_) {
    float mean = stats1[tid] * (1.f / (float)P1_);
    float var  = stats1[C_ + tid] * (1.f / (float)P1_) - mean * mean;
    float inv  = rsqrtf(var + 1e-5f);
    float s    = g1[tid] * inv;
    sc[tid] = s; sh[tid] = be1[tid] - mean * s; sw2[tid] = w2[tid];
  }
  __syncthreads();

  int t = blockIdx.x * 256 + tid;                  // (b,n)
  int b = t / N_, n = t - b * N_;
  size_t pbase = (size_t)t * K_;

  float acc[8];
  float bias2 = b2[0];
#pragma unroll
  for (int k = 0; k < 8; ++k) acc[k] = bias2;

  for (int c = 0; c < C_; ++c) {                   // 8 contiguous h values per load group
    const float* hp = hbuf + (size_t)c * P1_ + pbase;
    float s = sc[c], o = sh[c], wc = sw2[c];
#pragma unroll
    for (int k = 0; k < 8; ++k) {
      float hn = fmaxf(fmaf(hp[k], s, o), 0.f);
      acc[k] = fmaf(hn, wc, acc[k]);
    }
  }

  int id[8]; float w8[8]; float wsum = 1e-8f;
#pragma unroll
  for (int k = 0; k < 8; ++k) {
    float w = 1.f / (1.f + __expf(-acc[k]));
    w8[k] = w; wsum += w;
    out_w[pbase + k] = w;
    id[k] = idxw[pbase + k];
  }
  float inv = 1.f / wsum;

  for (int c = 0; c < C_; ++c) {
    const float* row = cf + ((size_t)b*C_ + c) * M_;
    float a = 0.f;
#pragma unroll
    for (int k = 0; k < 8; ++k) a = fmaf(row[id[k]], w8[k], a);
    interp[((size_t)b*C_ + c) * N_ + n] = a * inv;
  }
}

// ---------------------------------------------------------------- final MLP: out = Wm @ [interp;pf] + bm
__global__ __launch_bounds__(256) void gemm2_k(const float* __restrict__ pf,
                                               const float* __restrict__ interp,
                                               const float* __restrict__ mw,
                                               const float* __restrict__ mb,
                                               float* __restrict__ outp,
                                               float* __restrict__ part2) {
  __shared__ __bf16 lw[OUT_ * 2*C_];           // 128x128 bf16
  __shared__ float lsum[8][OUT_], lsq[8][OUT_];
  int tid = threadIdx.x;
  for (int i = tid; i < OUT_*2*C_; i += 256) lw[i] = (__bf16)mw[i];
  __syncthreads();

  const int wave = tid >> 5, lane = tid & 31;
  const int p = (blockIdx.x * 8 + wave) * 16 + (lane & 15);
  const int b = p / N_, n = p - b * N_;
  const int coff  = (lane >> 4) << 4;
  const int half8 = (lane >> 4) << 3;
  const int ml    = lane & 15;

  v8f acc[8];
#pragma unroll
  for (int t = 0; t < 8; ++t)
#pragma unroll
    for (int r = 0; r < 8; ++r) acc[t][r] = 0.f;

#pragma unroll
  for (int s = 0; s < 4; ++s) {
    v16bf bfrag;
#pragma unroll
    for (int j = 0; j < 16; ++j) {
      int c = s*32 + half8 + (j & 7) + ((j >> 3) << 4);
      float v = (c < C_) ? interp[((size_t)b*C_ + c      ) * N_ + n]
                         : pf    [((size_t)b*C_ + (c-C_)) * N_ + n];
      bfrag[j] = (__bf16)v;
    }
#pragma unroll
    for (int t = 0; t < 8; ++t) {
      v16bf afrag;
#pragma unroll
      for (int j = 0; j < 16; ++j) {
        int i = j >> 1, pb = j & 1;
        int kl = ((i >> 2) << 4) + half8 + ((i & 3) << 1) + pb;
        afrag[j] = lw[(t*16 + ml) * (2*C_) + s*32 + kl];
      }
      acc[t] = __builtin_amdgcn_wmma_f32_16x16x32_bf16(
          false, afrag, false, bfrag, (short)0, acc[t], false, false);
    }
  }

#pragma unroll
  for (int t = 0; t < 8; ++t) {
#pragma unroll
    for (int r = 0; r < 8; ++r) {
      int o = t*16 + half8 + r;
      float v = acc[t][r] + mb[o];
      outp[((size_t)b*OUT_ + o) * N_ + n] = v;
      float s1 = v, s2 = v*v;
#pragma unroll
      for (int m = 1; m < 16; m <<= 1) { s1 += __shfl_xor(s1, m, 32); s2 += __shfl_xor(s2, m, 32); }
      if (ml == 0) { lsum[wave][o] = s1; lsq[wave][o] = s2; }
    }
  }
  __syncthreads();
  if (tid < OUT_) {
    float a = 0.f, q = 0.f;
#pragma unroll
    for (int w = 0; w < 8; ++w) { a += lsum[w][tid]; q += lsq[w][tid]; }
    part2[(size_t)blockIdx.x * (2*OUT_) + tid]        = a;
    part2[(size_t)blockIdx.x * (2*OUT_) + OUT_ + tid] = q;
  }
}

// ---------------------------------------------------------------- final BN + ReLU (in place)
__global__ __launch_bounds__(256) void finbn_k(float* __restrict__ outp,
                                               const float* __restrict__ stats2,
                                               const float* __restrict__ g,
                                               const float* __restrict__ be) {
  size_t i = (size_t)blockIdx.x * 256 + threadIdx.x;
  int o = (int)((i / N_) % OUT_);
  float mean = stats2[o] * (1.f / (float)P2_);
  float var  = stats2[OUT_ + o] * (1.f / (float)P2_) - mean * mean;
  float inv  = rsqrtf(var + 1e-5f);
  float s    = g[o] * inv;
  float x    = outp[i];
  outp[i] = fmaxf(fmaf(x, s, be[o] - mean * s), 0.f);
}

__global__ __launch_bounds__(256) void copy_k(const float* __restrict__ src,
                                              float* __restrict__ dst) {
  size_t i = (size_t)blockIdx.x * 256 + threadIdx.x;
  dst[i] = src[i];
}

// ---------------------------------------------------------------- launch
extern "C" void kernel_launch(void* const* d_in, const int* in_sizes, int n_in,
                              void* d_out, int out_size, void* d_ws, size_t ws_size,
                              hipStream_t stream) {
  (void)in_sizes; (void)n_in; (void)out_size; (void)ws_size;
  const float* pc  = (const float*)d_in[0];   // points_coords  [B,3,N]
  const float* cc  = (const float*)d_in[1];   // centers_coords [B,3,M]
  const float* cf  = (const float*)d_in[2];   // centers_features [B,C,M]
  const float* pf  = (const float*)d_in[3];   // points_features  [B,C,N]
  const float* w1  = (const float*)d_in[4];
  const float* b1  = (const float*)d_in[5];
  const float* g1  = (const float*)d_in[6];
  const float* be1 = (const float*)d_in[7];
  const float* w2  = (const float*)d_in[8];
  const float* b2  = (const float*)d_in[9];
  const float* mw  = (const float*)d_in[10];
  const float* mb  = (const float*)d_in[11];
  const float* mg  = (const float*)d_in[12];
  const float* mbe = (const float*)d_in[13];

  // workspace layout
  char* ws = (char*)d_ws;
  float* packedC = (float*)ws;  ws += (size_t)B_*M_*4 * sizeof(float);        // 256 KB
  int*   idxw    = (int*)ws;    ws += (size_t)B_*N_*K_ * sizeof(int);         // 2 MB
  float* stats1  = (float*)ws;  ws += (size_t)2*C_ * sizeof(float);
  float* stats2  = (float*)ws;  ws += (size_t)2*OUT_ * sizeof(float);
  float* part1   = (float*)ws;  ws += (size_t)G1_BLOCKS * 2*C_ * sizeof(float);   // 2 MB
  float* part2   = (float*)ws;  ws += (size_t)G2_BLOCKS * 2*OUT_ * sizeof(float); // 512 KB
  float* interp  = (float*)ws;  ws += (size_t)B_*C_*N_ * sizeof(float);       // 16 MB
  float* hbuf    = (float*)ws;  ws += (size_t)C_*P1_ * sizeof(float);         // 128 MB

  // output layout: out | points_coords | idx | w
  float* out_o   = (float*)d_out;
  float* out_pc  = out_o  + (size_t)B_*OUT_*N_;
  float* out_idx = out_pc + (size_t)B_*3*N_;
  float* out_w   = out_idx + (size_t)B_*N_*K_;

  pack_centers_k<<<(B_*M_)/256, 256, 0, stream>>>(cc, packedC);
  knn_k<<<(B_*N_/16)/8, 256, (size_t)M_*4*sizeof(float), stream>>>(pc, packedC, idxw, out_idx);
  gemm1_k<<<G1_BLOCKS, 256, 0, stream>>>(pf, w1, b1, idxw, hbuf, part1);
  reduce_stats_k<<<1, 2*C_, 0, stream>>>(part1, stats1, G1_BLOCKS, 2*C_);
  wint_k<<<P2_/256, 256, 0, stream>>>(hbuf, stats1, g1, be1, w2, b2, cf, idxw, out_w, interp);
  gemm2_k<<<G2_BLOCKS, 256, 0, stream>>>(pf, interp, mw, mb, out_o, part2);
  reduce_stats_k<<<1, 2*OUT_, 0, stream>>>(part2, stats2, G2_BLOCKS, 2*OUT_);
  finbn_k<<<(B_*OUT_*N_)/256, 256, 0, stream>>>(out_o, stats2, mg, mbe);
  copy_k<<<(B_*3*N_)/256, 256, 0, stream>>>(pc, out_pc);
}
